// Encoder_32796370272476
// MI455X (gfx1250) — compile-verified, matrix operand order checked by reading, not run on hardware
//
#include <hip/hip_runtime.h>

// ---------------------------------------------------------------------------
// 2-layer GCN for MI455X (gfx1250): bf16 WMMA GEMMs + atomic edge scatter.
// ---------------------------------------------------------------------------

typedef __attribute__((ext_vector_type(16))) __bf16 v16bf;
typedef __attribute__((ext_vector_type(8)))  __bf16 v8bf;
typedef __attribute__((ext_vector_type(8)))  float  v8f;

#define IN_CH  256
#define HIDDEN 128
#define OUT_CH 64

// ---------------------------- degree / norm --------------------------------

__global__ void fill_ones_kernel(float* __restrict__ p, int n) {
  int i = blockIdx.x * blockDim.x + threadIdx.x;
  if (i < n) p[i] = 1.0f;   // self-loop contributes 1 to every node's degree
}

__global__ void deg_scatter_kernel(const long long* __restrict__ dst,
                                   float* __restrict__ deg, int E) {
  int i = blockIdx.x * blockDim.x + threadIdx.x;
  if (i < E) atomicAdd(&deg[(int)dst[i]], 1.0f);
}

__global__ void rsqrt_kernel(float* __restrict__ p, int n) {
  int i = blockIdx.x * blockDim.x + threadIdx.x;
  if (i < n) {
    float d = p[i];
    p[i] = (d > 0.0f) ? rsqrtf(d) : 0.0f;
  }
}

// ------------------- weight convert + transpose (f32 -> bf16) --------------
// Wt[n][k] = (bf16) W[k][n]  so B fragments are contiguous along K.

__global__ void convert_transpose_kernel(const float* __restrict__ W,
                                         __bf16* __restrict__ Wt,
                                         int K, int NOUT) {
  int i = blockIdx.x * blockDim.x + threadIdx.x;
  if (i < K * NOUT) {
    int k = i / NOUT;
    int n = i % NOUT;
    Wt[(size_t)n * K + k] = (__bf16)W[i];
  }
}

// ------------------------------ WMMA GEMM ----------------------------------
// out[M x NOUT] = A[M x K] * W[K x NOUT], A f32 (staged to LDS as bf16),
// W pre-transposed bf16 [NOUT][K]. One block per 16-row M tile; one wave per
// 16-col N tile. M must be a multiple of 16 (100000 = 6250*16).

template <int K, int NOUT>
__global__ void gemm_wmma_kernel(const float* __restrict__ A,
                                 const __bf16* __restrict__ Wt,
                                 float* __restrict__ out, int M) {
  __shared__ __bf16 sA[16][K];
  const int m0 = blockIdx.x * 16;

  // cooperative stage: 16 x K f32 -> bf16 in LDS (A read exactly once)
  for (int idx = threadIdx.x; idx < 16 * K; idx += blockDim.x) {
    int r = idx / K, c = idx % K;
    sA[r][c] = (__bf16)A[(size_t)(m0 + r) * K + c];
  }
  __syncthreads();

  const int wave  = threadIdx.x >> 5;
  const int lane  = threadIdx.x & 31;
  const int l15   = lane & 15;          // A row within tile / B-C column
  const int kb    = (lane >> 4) * 8;    // K sub-offset for upper half-wave
  const int nbase = wave * 16;

  v8f c = {};
  const __bf16* wrow = Wt + (size_t)(nbase + l15) * K;

#pragma unroll
  for (int kk = 0; kk < K; kk += 32) {
    // A fragment: rows in lanes, K={kk+kb+0..7} and {kk+16+kb+0..7}
    v8bf alo = *(const v8bf*)(&sA[l15][kk + kb]);
    v8bf ahi = *(const v8bf*)(&sA[l15][kk + 16 + kb]);
    v16bf a = __builtin_shufflevector(alo, ahi,
        0,1,2,3,4,5,6,7,8,9,10,11,12,13,14,15);
    // B fragment: column l15, same K pattern (contiguous in transposed W)
    v8bf blo = *(const v8bf*)(wrow + kk + kb);
    v8bf bhi = *(const v8bf*)(wrow + kk + 16 + kb);
    v16bf b = __builtin_shufflevector(blo, bhi,
        0,1,2,3,4,5,6,7,8,9,10,11,12,13,14,15);
    c = __builtin_amdgcn_wmma_f32_16x16x32_bf16(
        /*neg_a=*/false, a, /*neg_b=*/false, b,
        /*c_mod=*/(short)0, c, /*reuse_a=*/false, /*reuse_b=*/false);
  }

  // C/D layout: VGPR r -> M = r + 8*(lane>=16), N = l15
  const int mrow = (lane >> 4) * 8;
#pragma unroll
  for (int r = 0; r < 8; ++r) {
    out[(size_t)(m0 + mrow + r) * NOUT + nbase + l15] = c[r];
  }
}

// ----------------------------- aggregation ---------------------------------
// agg[i] = dinv[i]^2 * h[i]   (self-loop term; also initializes the buffer)

template <int C>
__global__ void agg_init_kernel(float* __restrict__ agg,
                                const float* __restrict__ h,
                                const float* __restrict__ dinv, int N) {
  size_t i = blockIdx.x * (size_t)blockDim.x + threadIdx.x;
  size_t total = (size_t)N * (C / 4);
  if (i >= total) return;
  int node = (int)(i / (C / 4));
  int c4   = (int)(i % (C / 4)) * 4;
  float w = dinv[node];
  w = w * w;
  const float4 hv = *(const float4*)(h + (size_t)node * C + c4);
  float4 o = make_float4(hv.x * w, hv.y * w, hv.z * w, hv.w * w);
  *(float4*)(agg + (size_t)node * C + c4) = o;
}

// agg[dst] += dinv[src]*dinv[dst] * h[src]   per edge (atomic scatter)

template <int C>
__global__ void agg_edge_kernel(float* __restrict__ agg,
                                const float* __restrict__ h,
                                const float* __restrict__ dinv,
                                const long long* __restrict__ src,
                                const long long* __restrict__ dst, int E) {
  size_t i = blockIdx.x * (size_t)blockDim.x + threadIdx.x;
  size_t total = (size_t)E * (C / 4);
  if (i >= total) return;
  int e  = (int)(i / (C / 4));
  int c4 = (int)(i % (C / 4)) * 4;
  int s = (int)src[e];
  int d = (int)dst[e];
  float w = dinv[s] * dinv[d];
  const float4 hv = *(const float4*)(h + (size_t)s * C + c4);  // L2-resident
  float* base = agg + (size_t)d * C + c4;
  atomicAdd(base + 0, hv.x * w);
  atomicAdd(base + 1, hv.y * w);
  atomicAdd(base + 2, hv.z * w);
  atomicAdd(base + 3, hv.w * w);
}

template <int C, bool RELU>
__global__ void bias_act_kernel(float* __restrict__ agg,
                                const float* __restrict__ b, int N) {
  size_t i = blockIdx.x * (size_t)blockDim.x + threadIdx.x;
  size_t total = (size_t)N * C;
  if (i >= total) return;
  int c = (int)(i % C);
  float v = agg[i] + b[c];
  if (RELU) v = fmaxf(v, 0.0f);
  agg[i] = v;
}

// ------------------------------- driver ------------------------------------

extern "C" void kernel_launch(void* const* d_in, const int* in_sizes, int n_in,
                              void* d_out, int out_size, void* d_ws, size_t ws_size,
                              hipStream_t stream) {
  const float*     x  = (const float*)d_in[0];
  const long long* ei = (const long long*)d_in[1];
  const float*     W1 = (const float*)d_in[2];
  const float*     b1 = (const float*)d_in[3];
  const float*     W2 = (const float*)d_in[4];
  const float*     b2 = (const float*)d_in[5];

  const int N = in_sizes[0] / IN_CH;   // 100000
  const int E = in_sizes[1] / 2;       // 1600000
  const long long* srcIdx = ei;        // edge_index[0]
  const long long* dstIdx = ei + E;    // edge_index[1]
  float* out = (float*)d_out;

  // -------- workspace layout (256B aligned slices) --------
  auto align256 = [](size_t v) { return (v + 255) & ~(size_t)255; };
  char* ws = (char*)d_ws;
  size_t off = 0;
  float*  dinv = (float*)(ws + off);  off = align256(off + (size_t)N * 4);
  __bf16* w1t  = (__bf16*)(ws + off); off = align256(off + (size_t)IN_CH * HIDDEN * 2);
  __bf16* w2t  = (__bf16*)(ws + off); off = align256(off + (size_t)HIDDEN * OUT_CH * 2);
  float*  h1   = (float*)(ws + off);  off = align256(off + (size_t)N * HIDDEN * 4);
  float*  h1a  = (float*)(ws + off);  off = align256(off + (size_t)N * HIDDEN * 4);
  float*  h2   = h1;  // reuse: h1 is dead once h1a is built

  const int B = 256;

  // degrees + dinv (self-loops folded in via init to 1)
  fill_ones_kernel<<<(N + B - 1) / B, B, 0, stream>>>(dinv, N);
  deg_scatter_kernel<<<(E + B - 1) / B, B, 0, stream>>>(dstIdx, dinv, E);
  rsqrt_kernel<<<(N + B - 1) / B, B, 0, stream>>>(dinv, N);

  // weights -> bf16, K-major
  convert_transpose_kernel<<<(IN_CH * HIDDEN + B - 1) / B, B, 0, stream>>>(W1, w1t, IN_CH, HIDDEN);
  convert_transpose_kernel<<<(HIDDEN * OUT_CH + B - 1) / B, B, 0, stream>>>(W2, w2t, HIDDEN, OUT_CH);

  // ---- layer 1: h1 = x @ W1 (WMMA), aggregate, +b1, relu ----
  gemm_wmma_kernel<IN_CH, HIDDEN>
      <<<N / 16, (HIDDEN / 16) * 32, 0, stream>>>(x, w1t, h1, N);
  {
    size_t ti = (size_t)N * (HIDDEN / 4);
    agg_init_kernel<HIDDEN><<<(unsigned)((ti + B - 1) / B), B, 0, stream>>>(h1a, h1, dinv, N);
    size_t te = (size_t)E * (HIDDEN / 4);
    agg_edge_kernel<HIDDEN><<<(unsigned)((te + B - 1) / B), B, 0, stream>>>(h1a, h1, dinv, srcIdx, dstIdx, E);
    size_t tb = (size_t)N * HIDDEN;
    bias_act_kernel<HIDDEN, true><<<(unsigned)((tb + B - 1) / B), B, 0, stream>>>(h1a, b1, N);
  }

  // ---- layer 2: h2 = h1a @ W2 (WMMA), aggregate into d_out, +b2 ----
  gemm_wmma_kernel<HIDDEN, OUT_CH>
      <<<N / 16, (OUT_CH / 16) * 32, 0, stream>>>(h1a, w2t, h2, N);
  {
    size_t ti = (size_t)N * (OUT_CH / 4);
    agg_init_kernel<OUT_CH><<<(unsigned)((ti + B - 1) / B), B, 0, stream>>>(out, h2, dinv, N);
    size_t te = (size_t)E * (OUT_CH / 4);
    agg_edge_kernel<OUT_CH><<<(unsigned)((te + B - 1) / B), B, 0, stream>>>(out, h2, dinv, srcIdx, dstIdx, E);
    size_t tb = (size_t)N * OUT_CH;
    bias_act_kernel<OUT_CH, false><<<(unsigned)((tb + B - 1) / B), B, 0, stream>>>(out, b2, N);
  }

  (void)n_in; (void)out_size; (void)ws_size;
}
